// VllmQuantLinearInt8_7078106103814
// MI455X (gfx1250) — 8-bit matrix operand order repaired and checked by reading, compiled, not run on hardware
//
#include <hip/hip_runtime.h>
#include <cstdint>

#define M_DIM 8192
#define K_DIM 4096
#define N_DIM 12288

#define BM 128
#define BN 128
#define BK 64
#define LDSROW 80

#define AS1 __attribute__((address_space(1)))
#define AS3 __attribute__((address_space(3)))

#define USE_ASYNC 1

typedef int v8i __attribute__((ext_vector_type(8)));
typedef int v4i __attribute__((ext_vector_type(4)));
typedef int v2i __attribute__((ext_vector_type(2)));
typedef int g4i __attribute__((vector_size(16)));

union FragA { v8i v; v2i d[4]; };
union FragB { v8i v; v4i q[2]; };

__global__ __launch_bounds__(256) void quant_rows(const float* __restrict__ x,
                                                  int8_t* __restrict__ q,
                                                  float* __restrict__ ascale) {
  const int row = blockIdx.x;
  const int tid = threadIdx.x;

  const float4* xp = (const float4*)(x + (size_t)row * K_DIM);
  float4 v[4];
  float amax = 0.0f;
#pragma unroll
  for (int i = 0; i < 4; ++i) {
    v[i] = xp[tid * 4 + i];
    amax = fmaxf(amax, fmaxf(fmaxf(fabsf(v[i].x), fabsf(v[i].y)),
                             fmaxf(fabsf(v[i].z), fabsf(v[i].w))));
  }
#pragma unroll
  for (int o = 16; o > 0; o >>= 1)
    amax = fmaxf(amax, __shfl_xor(amax, o, 32));

  __shared__ float smax[8];
  if ((tid & 31) == 0) smax[tid >> 5] = amax;
  __syncthreads();
  if (tid == 0) {
    float m = smax[0];
#pragma unroll
    for (int i = 1; i < 8; ++i) m = fmaxf(m, smax[i]);
    smax[0] = m;
  }
  __syncthreads();
  amax = fmaxf(smax[0], 1e-30f);
  const float inv = 127.0f / amax;

  uint32_t w[4];
#pragma unroll
  for (int i = 0; i < 4; ++i) {
    int a = max(-128, min(127, __float2int_rn(v[i].x * inv)));
    int b = max(-128, min(127, __float2int_rn(v[i].y * inv)));
    int c = max(-128, min(127, __float2int_rn(v[i].z * inv)));
    int d = max(-128, min(127, __float2int_rn(v[i].w * inv)));
    w[i] = (uint32_t)(a & 255) | ((uint32_t)(b & 255) << 8) |
           ((uint32_t)(c & 255) << 16) | ((uint32_t)(d & 255) << 24);
  }
  *(uint4*)(q + (size_t)row * K_DIM + tid * 16) = make_uint4(w[0], w[1], w[2], w[3]);
  if (tid == 0) ascale[row] = amax * (1.0f / 127.0f);
}

__global__ __launch_bounds__(256) void gemm_iu8(const int8_t* __restrict__ Aq,
                                                const int8_t* __restrict__ Wt,
                                                const float* __restrict__ ascale,
                                                const float* __restrict__ wscale,
                                                const float* __restrict__ bias,
                                                float* __restrict__ out) {
  const int tid  = threadIdx.x;
  const int wave = tid >> 5;
  const int lane = tid & 31;
  const int half = lane >> 4;
  const int r    = lane & 15;

  const int m0 = blockIdx.y * BM;
  const int n0 = blockIdx.x * BN;
  const int wm = wave >> 1;
  const int wn = wave & 1;

  const v8i vzero = {0, 0, 0, 0, 0, 0, 0, 0};
  v8i acc[2][4];
#pragma unroll
  for (int mi = 0; mi < 2; ++mi)
#pragma unroll
    for (int ni = 0; ni < 4; ++ni) acc[mi][ni] = vzero;

  constexpr int T = K_DIM / BK;

#if USE_ASYNC
  __shared__ __attribute__((aligned(16))) unsigned char As[2][BM * LDSROW];
  __shared__ __attribute__((aligned(16))) unsigned char Bs[2][BN * LDSROW];

  const int rowA      = tid >> 2;
  const int offA      = (tid & 3) << 4;
  const int8_t* gA_lo = Aq + (size_t)(m0 + rowA) * K_DIM + offA;
  const int8_t* gA_hi = gA_lo + (size_t)64 * K_DIM;
  const int8_t* gB_lo = Wt + (size_t)(n0 + rowA) * K_DIM + offA;
  const int8_t* gB_hi = gB_lo + (size_t)64 * K_DIM;
  const unsigned l_lo = rowA * LDSROW + offA;
  const unsigned l_hi = l_lo + 64 * LDSROW;

  auto issue = [&](int t) {
    const int buf = t & 1;
    const int k   = t * BK;
    __builtin_amdgcn_global_load_async_to_lds_b128(
        (AS1 g4i*)(gA_lo + k), (AS3 g4i*)(&As[buf][l_lo]), 0, 0);
    __builtin_amdgcn_global_load_async_to_lds_b128(
        (AS1 g4i*)(gA_hi + k), (AS3 g4i*)(&As[buf][l_hi]), 0, 0);
    __builtin_amdgcn_global_load_async_to_lds_b128(
        (AS1 g4i*)(gB_lo + k), (AS3 g4i*)(&Bs[buf][l_lo]), 0, 0);
    __builtin_amdgcn_global_load_async_to_lds_b128(
        (AS1 g4i*)(gB_hi + k), (AS3 g4i*)(&Bs[buf][l_hi]), 0, 0);
  };

  issue(0);
  for (int t = 0; t < T; ++t) {
    const int buf = t & 1;
    if (t + 1 < T) {
      issue(t + 1);
      __builtin_amdgcn_s_wait_asynccnt(4);
    } else {
      __builtin_amdgcn_s_wait_asynccnt(0);
    }
    __syncthreads();

    const unsigned char* Ab = As[buf];
    const unsigned char* Bb = Bs[buf];
#else
  __shared__ __attribute__((aligned(16))) unsigned char AsS[BM * LDSROW];
  __shared__ __attribute__((aligned(16))) unsigned char BsS[BN * LDSROW];
  for (int t = 0; t < T; ++t) {
    const int k0 = t * BK;
#pragma unroll
    for (int s = 0; s < 2; ++s) {
      const int seg = tid + s * 256;
      const int row = seg >> 2;
      const int off = (seg & 3) << 4;
      *(v4i*)(AsS + row * LDSROW + off) =
          *(const v4i*)(Aq + (size_t)(m0 + row) * K_DIM + k0 + off);
      *(v4i*)(BsS + row * LDSROW + off) =
          *(const v4i*)(Wt + (size_t)(n0 + row) * K_DIM + k0 + off);
      __builtin_prefetch(Aq + (size_t)(m0 + row) * K_DIM + k0 + BK + off, 0, 1);
      __builtin_prefetch(Wt + (size_t)(n0 + row) * K_DIM + k0 + BK + off, 0, 1);
    }
    __syncthreads();
    const unsigned char* Ab = AsS;
    const unsigned char* Bb = BsS;
#endif

    FragA fa[2];
#pragma unroll
    for (int mi = 0; mi < 2; ++mi) {
      const unsigned char* p = Ab + (wm * 32 + mi * 16 + r) * LDSROW + half * 8;
#pragma unroll
      for (int c = 0; c < 4; ++c) fa[mi].d[c] = *(const v2i*)(p + c * 16);
    }
    FragB fb[4];
#pragma unroll
    for (int ni = 0; ni < 4; ++ni) {
      const unsigned char* p = Bb + (wn * 64 + ni * 16 + r) * LDSROW + half * 8;
      const v2i b0 = *(const v2i*)(p), b1 = *(const v2i*)(p + 16);
      const v2i b2 = *(const v2i*)(p + 32), b3 = *(const v2i*)(p + 48);
      fb[ni].q[0] = (v4i){b0.x, b0.y, b1.x, b1.y};
      fb[ni].q[1] = (v4i){b2.x, b2.y, b3.x, b3.y};
    }

#pragma unroll
    for (int mi = 0; mi < 2; ++mi)
#pragma unroll
      for (int ni = 0; ni < 4; ++ni)
        acc[mi][ni] = __builtin_amdgcn_wmma_i32_16x16x64_iu8(
            true, fa[mi].v, true, fb[ni].v, acc[mi][ni], false, false);

    __syncthreads();
  }

#pragma unroll
  for (int mi = 0; mi < 2; ++mi) {
    const int mbase = m0 + wm * 32 + mi * 16 + half * 8;
    float as[8];
#pragma unroll
    for (int e = 0; e < 8; ++e) as[e] = ascale[mbase + e];
#pragma unroll
    for (int ni = 0; ni < 4; ++ni) {
      const int ncol = n0 + wn * 64 + ni * 16 + r;
      const float wsc = wscale[ncol];
      const float bb  = bias[ncol];
#pragma unroll
      for (int e = 0; e < 8; ++e) {
        out[(size_t)(mbase + e) * N_DIM + ncol] =
            (float)acc[mi][ni][e] * as[e] * wsc + bb;
      }
    }
  }
}

extern "C" void kernel_launch(void* const* d_in, const int* in_sizes, int n_in,
                              void* d_out, int out_size, void* d_ws, size_t ws_size,
                              hipStream_t stream) {
  (void)in_sizes; (void)n_in; (void)out_size; (void)ws_size;

  const float*  x      = (const float*)d_in[0];
  const int8_t* weight = (const int8_t*)d_in[1];
  const float*  wscale = (const float*)d_in[2];
  const float*  bias   = (const float*)d_in[3];
  float*        out    = (float*)d_out;

  float*  ascale = (float*)d_ws;
  int8_t* q      = (int8_t*)d_ws + (size_t)M_DIM * sizeof(float);

  quant_rows<<<M_DIM, 256, 0, stream>>>(x, q, ascale);

  dim3 grid(N_DIM / BN, M_DIM / BM);
  gemm_iu8<<<grid, 256, 0, stream>>>(q, weight, ascale, wscale, bias, out);
}
